// LinearInt4_9062380994905
// MI455X (gfx1250) — matrix operand order repaired and checked by reading, not compiled, not run on hardware
//
#include <hip/hip_runtime.h>
#include <stdint.h>

typedef __attribute__((ext_vector_type(16))) _Float16 v16h;
typedef __attribute__((ext_vector_type(8)))  float    v8f;
typedef __attribute__((ext_vector_type(8)))  int      v8i;
typedef __attribute__((ext_vector_type(4)))  int      v4i;
typedef __attribute__((ext_vector_type(4)))  float    v4f;

#define M_DIM    32
#define K_DIM    4096
#define N_DIM    11008
#define GSIZE    128
#define NWAVES   4
#define BLOCK_N  (NWAVES * 16)        // 64 columns per block
#define NTHREADS (NWAVES * 32)        // 128 threads, wave32
#define XS_STRIDE 136                 // halves per x row (128 + 8 pad, 272B)
#define KCHUNK   64                   // k-ints staged per async chunk per column
#define QSTRIDE  68                   // ints per column in LDS (64 + 4 pad, 272B)
#define QBUF_INTS (16 * QSTRIDE)      // per wave, per buffer

union U16h { v16h h; v8i i; };

__global__ __launch_bounds__(NTHREADS)
void int4_gemm_wmma(const float* __restrict__ x,
                    const int*   __restrict__ qw,
                    const float* __restrict__ sz,
                    float*       __restrict__ out) {
  __shared__ __attribute__((aligned(16))) _Float16 xs[M_DIM * XS_STRIDE];
  __shared__ __attribute__((aligned(16))) int qbuf[NWAVES * 2 * QBUF_INTS];

  const int tid   = threadIdx.x;
  const int lane  = tid & 31;
  const int wave  = tid >> 5;
  const int lhalf = lane & 15;           // column-in-tile / row-in-tile
  const int khalf = (lane >> 4) << 3;    // 0 or 8 : wave32 16-bit K sub-block

  const int n0 = blockIdx.x * BLOCK_N + wave * 16;  // wave's first column
  const int n  = n0 + lhalf;                        // this lane's column

  int* qb = &qbuf[wave * 2 * QBUF_INTS];            // wave-private double buffer
  const unsigned qlds0 = (unsigned)(uintptr_t)qb;   // LDS byte offset (low 32b)

  // Issue async global->LDS copy of weight chunk c into buffer `buf`.
  // Geometry: 16 cols x 64 ints = 256 x 16B units; 32 lanes x 8 instructions.
  auto issue_async = [&](int c, int buf) {
    const unsigned base = qlds0 + (unsigned)buf * (QBUF_INTS * 4u);
    #pragma unroll
    for (int j = 0; j < 8; ++j) {
      const int li  = lane + 32 * j;     // 16B-unit index
      const int col = li >> 4;           // 0..15
      const int u   = li & 15;           // 0..15 (16B units within column)
      const int* gp = qw + (long)(n0 + col) * K_DIM + c * KCHUNK + u * 4;
      const unsigned ldsa = base + (unsigned)(col * QSTRIDE + u * 4) * 4u;
      asm volatile("global_load_async_to_lds_b128 %0, %1, off"
                   :: "v"(ldsa), "v"((uint64_t)(uintptr_t)gp)
                   : "memory");
    }
  };

  v8f c0 = {};   // M rows 0..15
  v8f c1 = {};   // M rows 16..31

  issue_async(0, 0);                     // prologue prefetch

  float scale = 0.0f, bofs = 0.0f;

  for (int c = 0; c < K_DIM / KCHUNK; ++c) {
    if ((c & 1) == 0) {
      const int kg    = c >> 1;
      const int kbase = kg * GSIZE;
      // ---- cooperatively stage x[0:32, kbase:kbase+128] as f16 into LDS ----
      __syncthreads();
      #pragma unroll
      for (int it = 0; it < 8; ++it) {
        const int fi = tid + it * NTHREADS;   // float4 index 0..1023
        const int m  = fi >> 5;
        const int kl = (fi & 31) << 2;
        const v4f v  = *(const v4f*)(x + (long)m * K_DIM + kbase + kl);
        _Float16* d  = &xs[m * XS_STRIDE + kl];
        d[0] = (_Float16)v.x; d[1] = (_Float16)v.y;
        d[2] = (_Float16)v.z; d[3] = (_Float16)v.w;
      }
      __syncthreads();
      // per-group dequant constants: w = (q-8)*s + z == q*s + (z - 8s)
      scale = sz[((long)kg * N_DIM + n) * 2 + 0];
      const float zero = sz[((long)kg * N_DIM + n) * 2 + 1];
      bofs = zero - 8.0f * scale;
    }

    // wait for this chunk's async weights, then prefetch the next chunk
    asm volatile("s_wait_asynccnt 0x0" ::: "memory");
    if (c + 1 < K_DIM / KCHUNK) issue_async(c + 1, (c + 1) & 1);

    const int* qp = (const int*)(qb + (c & 1) * QBUF_INTS) + lhalf * QSTRIDE;

    #pragma unroll
    for (int kk = 0; kk < 2; ++kk) {
      const int ka = (c & 1) * 64 + kk * 32 + khalf;   // k within xs window
      const int kb = kk * 32 + khalf;                  // k within weight chunk

      // ---- A fragments from LDS (row-major matches 16-bit A layout) ----
      v16h a0, a1;
      {
        const v4i* r0 = (const v4i*)&xs[lhalf * XS_STRIDE + ka];
        const v4i* r1 = (const v4i*)&xs[(lhalf + 16) * XS_STRIDE + ka];
        const v4i l0 = r0[0], h0 = r0[2];
        const v4i l1 = r1[0], h1 = r1[2];
        U16h u0, u1;
        #pragma unroll
        for (int i = 0; i < 4; ++i) {
          u0.i[i] = l0[i]; u0.i[i + 4] = h0[i];
          u1.i[i] = l1[i]; u1.i[i + 4] = h1[i];
        }
        a0 = u0.h; a1 = u1.h;
      }

      // ---- B fragment: dequantize int4-in-int32 weights from LDS ----
      const v4i q0 = *(const v4i*)(qp + kb);
      const v4i q1 = *(const v4i*)(qp + kb + 4);
      const v4i q2 = *(const v4i*)(qp + kb + 16);
      const v4i q3 = *(const v4i*)(qp + kb + 20);
      v16h b;
      #pragma unroll
      for (int i = 0; i < 4; ++i) {
        b[i]      = (_Float16)((float)q0[i] * scale + bofs);
        b[i + 4]  = (_Float16)((float)q1[i] * scale + bofs);
        b[i + 8]  = (_Float16)((float)q2[i] * scale + bofs);
        b[i + 12] = (_Float16)((float)q3[i] * scale + bofs);
      }

      c0 = __builtin_amdgcn_wmma_f32_16x16x32_f16(false, a0, false, b,
                                                  (short)0, c0, false, false);
      c1 = __builtin_amdgcn_wmma_f32_16x16x32_f16(false, a1, false, b,
                                                  (short)0, c1, false, false);
    }
  }

  // ---- store C: lanes 0-15 -> M=j, lanes 16-31 -> M=8+j (f32 C layout) ----
  const int mofs = (lane >> 4) << 3;
  #pragma unroll
  for (int j = 0; j < 8; ++j) {
    out[(long)(j + mofs)      * N_DIM + n] = c0[j];
    out[(long)(j + mofs + 16) * N_DIM + n] = c1[j];
  }
}

extern "C" void kernel_launch(void* const* d_in, const int* in_sizes, int n_in,
                              void* d_out, int out_size, void* d_ws, size_t ws_size,
                              hipStream_t stream) {
  const float* x  = (const float*)d_in[0];
  const int*   qw = (const int*)d_in[1];
  const float* sz = (const float*)d_in[2];
  float* out      = (float*)d_out;
  (void)in_sizes; (void)n_in; (void)out_size; (void)d_ws; (void)ws_size;

  dim3 grid(N_DIM / BLOCK_N);   // 11008 / 64 = 172 workgroups
  int4_gemm_wmma<<<grid, NTHREADS, 0, stream>>>(x, qw, sz, out);
}
